// DirectionalScanner_27376121545413
// MI455X (gfx1250) — matrix operand order repaired and checked by reading, not compiled, not run on hardware
//
#include <hip/hip_runtime.h>
#include <hip/hip_bf16.h>
#include <math.h>

// ---------------------------------------------------------------------------
// 4-direction GRU scanner for MI455X (gfx1250, wave32, WMMA bf16, async-LDS).
//
// Pipeline (all on `stream`):
//   k_prep  : fold Wi/W_ih into Wc=W_ih@Wi (192x192 bf16), bc; bf16 W_hh, Wo
//   k_gates : G[p,d,{r,z,n,pad}] = x[p,:] @ Wc^T + bc  (131072x192x192 GEMM)
//   k_scan  : 4096 independent GRU sequences (4 dirs x 1024), 16 seqs/wave,
//             24 v_wmma/step; gate fetch = 1 b128/lane-elem; per-direction H
//             buffers (pure stores -> bitwise deterministic, no atomics)
//   k_out   : out[p,c] = (mean_dir H) @ Wo^T + bo  (131072x192x64 GEMM)
//
// Weight tiles are staged global->LDS with GLOBAL_LOAD_ASYNC_TO_LDS_B128
// (ASYNCcnt + s_wait_asynccnt), overlapped with f32->bf16 conversion work.
// ~26 GFLOP total; G (134MB) and H (134MB) are L2-resident (192MB L2).
// Workspace requirement: ~257 MB.
// ---------------------------------------------------------------------------

typedef __attribute__((ext_vector_type(16))) __bf16 v16bf;
typedef __attribute__((ext_vector_type(8)))  __bf16 v8bf;
typedef __attribute__((ext_vector_type(8)))  float  v8f;
typedef __attribute__((ext_vector_type(4)))  float  v4f;

#define BB 8
#define CC 192
#define HH 128
#define WW 128
#define DD 64
#define GG 192
#define PIX (BB*HH*WW)          // 131072

// workspace byte offsets
#define WS_WC   0u               // 192*192 bf16  (73728)
#define WS_BC   73728u           // 192 f32
#define WS_WHH  74496u           // 192*64 bf16
#define WS_WO   99072u           // 192*64 bf16
#define WS_G    123648u          // PIX*64*4 f32 (r,z,n,pad)  (134217728)
#define WS_HS   134341376u       // 4 * PIX*64 f32            (134217728)
#define HS_DIR_STRIDE (PIX*64)   // 8388608 floats per direction

__device__ __forceinline__ __bf16 f2bf(float f) {
  unsigned u = __float_as_uint(f);
  u += 0x7FFFu + ((u >> 16) & 1u);           // round-to-nearest-even
  unsigned short s = (unsigned short)(u >> 16);
  __bf16 r; __builtin_memcpy(&r, &s, 2); return r;
}

__device__ __forceinline__ v8f vzero() {
  v8f z = {0.f,0.f,0.f,0.f,0.f,0.f,0.f,0.f}; return z;
}

__device__ __forceinline__ v8f wmma_bf(v16bf a, v16bf b, v8f c) {
  return __builtin_amdgcn_wmma_f32_16x16x32_bf16(false, a, false, b, (short)0, c,
                                                 false, false);
}

__device__ __forceinline__ float fsig(float x) { return 1.0f / (1.0f + __expf(-x)); }
__device__ __forceinline__ float ftanh(float x) {
#if __has_builtin(__builtin_amdgcn_tanhf)
  return __builtin_amdgcn_tanhf(x);           // gfx1250 V_TANH_F32
#else
  return tanhf(x);
#endif
}

// Async global->LDS copy, GVS mode: LDS[ldsAddr] = MEM[base + gOff], 16B/lane.
// Tracked by ASYNCcnt; fence with wait_async() before LDS is consumed.
__device__ __forceinline__ void async_copy_b128(unsigned ldsAddr, const void* base,
                                                unsigned gOff) {
  asm volatile("global_load_async_to_lds_b128 %0, %1, %2"
               :: "v"(ldsAddr), "v"(gOff), "s"(base) : "memory");
}
__device__ __forceinline__ void wait_async() {
  asm volatile("s_wait_asynccnt 0" ::: "memory");
}
__device__ __forceinline__ unsigned lds_off(const void* p) {
  return (unsigned)(uintptr_t)p;   // generic LDS pointer: low 32 bits = LDS offset
}

// A fragment (16 rows x 32 K, bf16): rows m=lane&15; lanes 0-15 hold
// K=[0..7],[16..23]; lanes 16-31 K=[8..15],[24..31].  base is row-major [m][k].
__device__ __forceinline__ v16bf load_afrag(const __bf16* base, int strideH,
                                            int kBase, int lane) {
  int m = lane & 15, hi = lane >> 4;
  const __bf16* r = base + m * strideH + kBase + hi * 8;
  union { v16bf v; v8bf h[2]; } f;
  f.h[0] = *(const v8bf*)(r);
  f.h[1] = *(const v8bf*)(r + 16);
  return f.v;
}

// B fragment (32 K x 16 N): lane n&15 = column N; lanes 0-15 K=[0..7],[16..23]; 16-31 K=[8..15],[24..31].
// base is row-major [n][k] (K contiguous per output column).
__device__ __forceinline__ v16bf load_bfrag(const __bf16* base, int strideH,
                                            int kBase, int lane) {
  int n = lane & 15, hi = lane >> 4;
  const __bf16* r = base + n * strideH + kBase + hi * 8;
  union { v16bf v; v8bf h[2]; } f;
  f.h[0] = *(const v8bf*)(r);
  f.h[1] = *(const v8bf*)(r + 16);
  return f.v;
}

// ---------------------------------------------------------------------------
// k_prep: Wc[g,c] = sum_d W_ih[g,d]*Wi[d,c];  bc[g] = W_ih[g,:]@bi + b_ih[g];
//         bf16 copies of W_hh [g][d] and Wo [c][d].
// ---------------------------------------------------------------------------
__global__ __launch_bounds__(256) void k_prep(const float* __restrict__ Wi,
                                              const float* __restrict__ bi,
                                              const float* __restrict__ W_ih,
                                              const float* __restrict__ W_hh,
                                              const float* __restrict__ b_ih,
                                              const float* __restrict__ Wo,
                                              __bf16* __restrict__ WcBf,
                                              float* __restrict__ bc,
                                              __bf16* __restrict__ WhhBf,
                                              __bf16* __restrict__ WoBf) {
  int tid = threadIdx.x;
  for (int i = tid; i < GG * CC; i += 256) {
    int g = i / CC, c = i % CC;
    float s = 0.f;
    #pragma unroll 8
    for (int d = 0; d < DD; ++d) s += W_ih[g * DD + d] * Wi[d * CC + c];
    WcBf[i] = f2bf(s);
  }
  for (int g = tid; g < GG; g += 256) {
    float s = b_ih[g];
    #pragma unroll 8
    for (int d = 0; d < DD; ++d) s += W_ih[g * DD + d] * bi[d];
    bc[g] = s;
  }
  for (int i = tid; i < GG * DD; i += 256) WhhBf[i] = f2bf(W_hh[i]);
  for (int i = tid; i < CC * DD; i += 256) WoBf[i]  = f2bf(Wo[i]);
}

// ---------------------------------------------------------------------------
// k_gates: one workgroup per (b,h) strip; M=w (128, 16/wave), N=g (192), K=c
// (192, split in two 96-halves to stay under 64KB static LDS).  Wc tiles are
// staged by async-LDS DMA, overlapped with the f32->bf16 x-strip conversion.
// Output layout: G[pixel][d][4] = {r,z,n,pad} so the scan reads one b128.
// ---------------------------------------------------------------------------
#define XS_S 96     // xs row stride (halves): 192B rows, 16B-aligned chunks
#define WC_S 104    // WcL row stride (halves): 208B rows, 16-bank spread
__global__ __launch_bounds__(256) void k_gates(const float* __restrict__ x,
                                               const __bf16* __restrict__ WcBf,
                                               const float* __restrict__ bc,
                                               float* __restrict__ G) {
  __shared__ __bf16 xsL[128 * XS_S];   // 24576 B
  __shared__ __bf16 WcL[192 * WC_S];   // 39936 B
  int tid = threadIdx.x, lane = tid & 31, wave = tid >> 5;
  int bh = blockIdx.x;
  int b = bh >> 7, h = bh & 127;
  const float* xb = x + (size_t)b * CC * HH * WW + (size_t)h * WW;  // + c*16384 + w
  float* Gp = G + (size_t)bh * 128 * 256;   // pixel p = bh*128+w, 256 f32/pixel
  int n15 = lane & 15, hi = lane >> 4;
  int w0 = wave * 16;
  unsigned wcBase = lds_off(&WcL[0]);

  v8f acc[12];
  #pragma unroll
  for (int i = 0; i < 12; ++i) acc[i] = vzero();

  for (int kh = 0; kh < 2; ++kh) {
    int cB = kh * 96;
    __syncthreads();
    // async DMA: Wc columns cB..cB+95 for all 192 rows (192*12 b128 chunks)
    for (int i = tid; i < 192 * 12; i += 256) {
      int g = i / 12, k = (i % 12) * 16;           // k: byte offset within row
      async_copy_b128(wcBase + (unsigned)(g * (WC_S * 2) + k), WcBf,
                      (unsigned)(g * (CC * 2) + cB * 2 + k));
    }
    // meanwhile: stage x strip (96 c x 128 w) as bf16, row-major [w][c]
    for (int i = tid; i < 96 * 128; i += 256) {
      int c = i >> 7, w = i & 127;
      xsL[w * XS_S + c] = f2bf(xb[(size_t)(cB + c) * (HH * WW) + w]);
    }
    wait_async();
    __syncthreads();

    v16bf aF0 = load_afrag(&xsL[w0 * XS_S], XS_S,  0, lane);
    v16bf aF1 = load_afrag(&xsL[w0 * XS_S], XS_S, 32, lane);
    v16bf aF2 = load_afrag(&xsL[w0 * XS_S], XS_S, 64, lane);
    #pragma unroll
    for (int nt = 0; nt < 12; ++nt) {
      const __bf16* bb = &WcL[nt * 16 * WC_S];
      v16bf b0 = load_bfrag(bb, WC_S,  0, lane);
      v16bf b1 = load_bfrag(bb, WC_S, 32, lane);
      v16bf b2 = load_bfrag(bb, WC_S, 64, lane);
      acc[nt] = wmma_bf(aF0, b0, acc[nt]);
      acc[nt] = wmma_bf(aF1, b1, acc[nt]);
      acc[nt] = wmma_bf(aF2, b2, acc[nt]);
    }
  }
  // Interleaved store: for d-chunk j, gates live in tiles j (r), j+4 (z), j+8 (n).
  // Each element becomes one b128 {r,z,n,0}; half-wave writes 256B contiguous.
  #pragma unroll
  for (int j = 0; j < 4; ++j) {
    float bR = bc[      j * 16 + n15];
    float bZ = bc[ 64 + j * 16 + n15];
    float bN = bc[128 + j * 16 + n15];
    #pragma unroll
    for (int e = 0; e < 8; ++e) {
      int w = w0 + e + hi * 8;
      v4f v = { acc[j][e] + bR, acc[j + 4][e] + bZ, acc[j + 8][e] + bN, 0.f };
      *(v4f*)(Gp + (size_t)w * 256 + (j * 16 + n15) * 4) = v;
    }
  }
}

// ---------------------------------------------------------------------------
// k_scan: 4096 sequences (dir-major), 16 per wave, 8 waves/WG -> 32 WGs.
// Per step: hg = h @ W_hh^T via 24 v_wmma_f32_16x16x32_bf16; GRU gate math
// (v_exp/v_tanh); h written back to per-wave LDS staging (bf16) for the next
// step's A fragments (DS ops are in-order per wave -> no barrier needed).
// ---------------------------------------------------------------------------
#define WH_S 72     // W_hh LDS row stride (halves): 144B rows
#define HS_S 72     // h staging row stride (halves)
__global__ __launch_bounds__(256) void k_scan(const float* __restrict__ G,
                                              const __bf16* __restrict__ WhhBf,
                                              const float* __restrict__ b_hh,
                                              float* __restrict__ Hs) {
  __shared__ __bf16 WhhL[192 * WH_S];        // 27648 B
  __shared__ __bf16 hseq[8 * 16 * HS_S];     // 18432 B
  int tid = threadIdx.x, lane = tid & 31, wave = tid >> 5;

  // async DMA: W_hh bf16 (192 rows x 128B) into padded LDS rows
  unsigned whBase = lds_off(&WhhL[0]);
  for (int i = tid; i < 192 * 8; i += 256) {
    int g = i >> 3, k = (i & 7) * 16;
    async_copy_b128(whBase + (unsigned)(g * (WH_S * 2) + k), WhhBf,
                    (unsigned)(g * (DD * 2) + k));
  }
  __bf16* hsq = &hseq[wave * 16 * HS_S];
  for (int i = lane; i < 16 * (HS_S / 2); i += 32)  // zero own staging (h0 = 0)
    ((unsigned*)hsq)[i] = 0u;
  wait_async();
  __syncthreads();

  int s16 = blockIdx.x * 8 + wave;       // 0..255
  int dir = s16 >> 6;                    // 0:right 1:down 2:left 3:up
  int rem = s16 & 63;
  int b  = rem >> 3;
  int a0 = (rem & 7) * 16;
  bool horiz = (dir == 0) || (dir == 2); // scan along w
  bool rev   = (dir >= 2);
  int mStep = horiz ? 128 : 1;
  int tStep = horiz ? 1 : 128;
  int sBase = b * (HH * WW) + a0 * (horiz ? 128 : 1);
  int n15 = lane & 15, hi = lane >> 4;

  int pB[8];
  #pragma unroll
  for (int e = 0; e < 8; ++e) pB[e] = sBase + (e + hi * 8) * mStep;

  float bh[3][4];                        // recurrent biases, per-lane constants
  #pragma unroll
  for (int q = 0; q < 3; ++q)
    #pragma unroll
    for (int j = 0; j < 4; ++j)
      bh[q][j] = b_hh[q * 64 + j * 16 + n15];

  v8f hst[4];
  #pragma unroll
  for (int j = 0; j < 4; ++j) hst[j] = vzero();

  float* Hd = Hs + (size_t)dir * HS_DIR_STRIDE;

  for (int t = 0; t < 128; ++t) {
    int tt = rev ? (127 - t) : t;
    int pT = tt * tStep;
    v16bf aF0 = load_afrag(hsq, HS_S,  0, lane);
    v16bf aF1 = load_afrag(hsq, HS_S, 32, lane);

    #pragma unroll
    for (int j = 0; j < 4; ++j) {                 // d-chunk of 16
      v8f aR = vzero(), aZ = vzero(), aN = vzero();
      {
        const __bf16* bb = &WhhL[(0 * 64 + j * 16) * WH_S];
        aR = wmma_bf(aF0, load_bfrag(bb, WH_S,  0, lane), aR);
        aR = wmma_bf(aF1, load_bfrag(bb, WH_S, 32, lane), aR);
      }
      {
        const __bf16* bb = &WhhL[(1 * 64 + j * 16) * WH_S];
        aZ = wmma_bf(aF0, load_bfrag(bb, WH_S,  0, lane), aZ);
        aZ = wmma_bf(aF1, load_bfrag(bb, WH_S, 32, lane), aZ);
      }
      {
        const __bf16* bb = &WhhL[(2 * 64 + j * 16) * WH_S];
        aN = wmma_bf(aF0, load_bfrag(bb, WH_S,  0, lane), aN);
        aN = wmma_bf(aF1, load_bfrag(bb, WH_S, 32, lane), aN);
      }
      int dB = j * 16 + n15;
      v8f hn;
      #pragma unroll
      for (int e = 0; e < 8; ++e) {
        size_t pix = (size_t)(pB[e] + pT);
        // one 16B load: {xr, xz, xn, pad} for (pixel, hidden-dim dB)
        v4f gv = *(const v4f*)(G + pix * 256 + dB * 4);
        float r  = fsig(gv[0] + aR[e] + bh[0][j]);
        float z  = fsig(gv[1] + aZ[e] + bh[1][j]);
        float nn = ftanh(gv[2] + r * (aN[e] + bh[2][j]));
        float hv = (1.0f - z) * nn + z * hst[j][e];
        hn[e] = hv;
        hsq[(e + hi * 8) * HS_S + dB] = f2bf(hv);      // next-step A operand
        Hd[pix * 64u + dB] = hv;                       // private per-dir buffer
      }
      hst[j] = hn;
    }
  }
}

// ---------------------------------------------------------------------------
// k_out: out[b,c,h,w] = (mean of 4 dir H)[p,:] @ Wo^T + bo.  One WG per (b,h);
// D tiles transposed through LDS so global stores are w-contiguous.
// ---------------------------------------------------------------------------
#define ST_S 20     // f32 stage row stride (conflict-spread, 16B aligned rows)
__global__ __launch_bounds__(256) void k_out(const float* __restrict__ Hs,
                                             const __bf16* __restrict__ WoBf,
                                             const float* __restrict__ bo,
                                             float* __restrict__ out) {
  __shared__ __bf16 WoL[192 * WH_S];       // 27648 B
  __shared__ __bf16 hb[128 * HS_S];        // 18432 B
  __shared__ float  stg[8][16 * ST_S];     // 10240 B
  int tid = threadIdx.x, lane = tid & 31, wave = tid >> 5;
  int bh = blockIdx.x;
  int b = bh >> 7, h = bh & 127;

  // async DMA: Wo bf16 (192 rows x 128B) into padded LDS rows
  unsigned woBase = lds_off(&WoL[0]);
  for (int i = tid; i < 192 * 8; i += 256) {
    int c = i >> 3, k = (i & 7) * 16;
    async_copy_b128(woBase + (unsigned)(c * (WH_S * 2) + k), WoBf,
                    (unsigned)(c * (DD * 2) + k));
  }
  // meanwhile: average 4 direction buffers, convert to bf16
  for (int i = tid; i < 128 * 64; i += 256) {
    int w = i >> 6, d = i & 63;
    size_t base = (size_t)bh * (128 * 64) + (size_t)w * 64 + d;
    float s = Hs[base] + Hs[base + (size_t)HS_DIR_STRIDE]
            + Hs[base + 2 * (size_t)HS_DIR_STRIDE]
            + Hs[base + 3 * (size_t)HS_DIR_STRIDE];
    hb[w * HS_S + d] = f2bf(0.25f * s);
  }
  wait_async();
  __syncthreads();

  int n15 = lane & 15, hi = lane >> 4;
  int w0 = wave * 16;
  float* ob = out + (size_t)b * CC * HH * WW + (size_t)h * WW;  // + c*16384 + w
  v16bf aF0 = load_afrag(&hb[w0 * HS_S], HS_S,  0, lane);
  v16bf aF1 = load_afrag(&hb[w0 * HS_S], HS_S, 32, lane);
  float* st = stg[wave];
  int cRow = lane >> 1, hf = lane & 1;

  #pragma unroll 1
  for (int nt = 0; nt < 12; ++nt) {
    const __bf16* bb = &WoL[nt * 16 * WH_S];
    v8f acc = vzero();
    acc = wmma_bf(aF0, load_bfrag(bb, WH_S,  0, lane), acc);
    acc = wmma_bf(aF1, load_bfrag(bb, WH_S, 32, lane), acc);
    float bias = bo[nt * 16 + n15];
    #pragma unroll
    for (int e = 0; e < 8; ++e)
      st[n15 * ST_S + e + hi * 8] = acc[e] + bias;   // stage as [c][w]
    // cooperative transposed store: 2 lanes per c-row, 32B contiguous each
    v4f v0 = *(const v4f*)&st[cRow * ST_S + hf * 8];
    v4f v1 = *(const v4f*)&st[cRow * ST_S + hf * 8 + 4];
    float* orow = ob + (size_t)(nt * 16 + cRow) * (HH * WW) + w0 + hf * 8;
    *(v4f*)orow       = v0;
    *(v4f*)(orow + 4) = v1;
  }
}

// ---------------------------------------------------------------------------
extern "C" void kernel_launch(void* const* d_in, const int* in_sizes, int n_in,
                              void* d_out, int out_size, void* d_ws, size_t ws_size,
                              hipStream_t stream) {
  const float* x    = (const float*)d_in[0];
  const float* Wi   = (const float*)d_in[1];
  const float* bi   = (const float*)d_in[2];
  const float* W_ih = (const float*)d_in[3];
  const float* W_hh = (const float*)d_in[4];
  const float* b_ih = (const float*)d_in[5];
  const float* b_hh = (const float*)d_in[6];
  const float* Wo   = (const float*)d_in[7];
  const float* bo   = (const float*)d_in[8];

  char* ws = (char*)d_ws;
  __bf16* WcBf  = (__bf16*)(ws + WS_WC);
  float*  bc    = (float*)(ws + WS_BC);
  __bf16* WhhBf = (__bf16*)(ws + WS_WHH);
  __bf16* WoBf  = (__bf16*)(ws + WS_WO);
  float*  G     = (float*)(ws + WS_G);
  float*  Hs    = (float*)(ws + WS_HS);
  float*  outp  = (float*)d_out;

  k_prep <<<1,    256, 0, stream>>>(Wi, bi, W_ih, W_hh, b_ih, Wo, WcBf, bc, WhhBf, WoBf);
  k_gates<<<1024, 256, 0, stream>>>(x, WcBf, bc, G);
  k_scan <<<32,   256, 0, stream>>>(G, WhhBf, b_hh, Hs);
  k_out  <<<1024, 256, 0, stream>>>(Hs, WoBf, bo, outp);
}
